// PUFM_59768764891781
// MI455X (gfx1250) — matrix operand order repaired and checked by reading, not compiled, not run on hardware
//
#include <hip/hip_runtime.h>

// ---------------------------------------------------------------------------
// Types
// ---------------------------------------------------------------------------
typedef _Float16 h16;
typedef __attribute__((ext_vector_type(16))) _Float16 v16h;
typedef __attribute__((ext_vector_type(8)))  _Float16 v8h;
typedef __attribute__((ext_vector_type(8)))  float    v8f;
typedef __attribute__((ext_vector_type(4)))  float    v4f;

// ---------------------------------------------------------------------------
// Geometry kernels
// ---------------------------------------------------------------------------

// (B,3,N) -> (B,N,3)
__global__ void k_xyz_t(const float* __restrict__ q, float* __restrict__ xyz, int N) {
  int i = blockIdx.x * blockDim.x + threadIdx.x;
  int b = blockIdx.y;
  if (i >= N) return;
  for (int d = 0; d < 3; ++d)
    xyz[((size_t)b * N + i) * 3 + d] = q[((size_t)b * 3 + d) * N + i];
}

// KNN-16 of the full cloud (cloud staged in LDS); writes l0_points feature
// buffer [B][N][Cs] f16, channel = dim*16 + k, channels >=48 zeroed.
__global__ void k_knn(const float* __restrict__ xyz, h16* __restrict__ feat0, int N, int Cs) {
  extern __shared__ float sxyz[];
  int b = blockIdx.y;
  const float* xb = xyz + (size_t)b * N * 3;
  for (int i = threadIdx.x; i < N * 3; i += blockDim.x) sxyz[i] = xb[i];
  __syncthreads();
  int n = blockIdx.x * blockDim.x + threadIdx.x;
  if (n >= N) return;
  float px = sxyz[n * 3 + 0], py = sxyz[n * 3 + 1], pz = sxyz[n * 3 + 2];
  float bd[16]; int bi[16];
  #pragma unroll
  for (int k = 0; k < 16; ++k) { bd[k] = 3.4e38f; bi[k] = 0; }
  for (int j = 0; j < N; ++j) {
    float dx = sxyz[j * 3 + 0] - px;
    float dy = sxyz[j * 3 + 1] - py;
    float dz = sxyz[j * 3 + 2] - pz;
    float d = dx * dx + dy * dy + dz * dz;
    if (d < bd[15]) {
      int k = 15;
      while (k > 0 && bd[k - 1] > d) { bd[k] = bd[k - 1]; bi[k] = bi[k - 1]; --k; }
      bd[k] = d; bi[k] = j;
    }
  }
  h16* out = feat0 + ((size_t)b * N + n) * Cs;
  for (int k = 0; k < 16; ++k) {
    int j = bi[k];
    out[0 * 16 + k] = (h16)sxyz[j * 3 + 0];
    out[1 * 16 + k] = (h16)sxyz[j * 3 + 1];
    out[2 * 16 + k] = (h16)sxyz[j * 3 + 2];
  }
  for (int c = 48; c < Cs; ++c) out[c] = (h16)0.0f;
}

// Farthest point sampling: one workgroup (1024 thr) per batch, sequential.
__global__ void k_fps(const float* __restrict__ xyz, int N, int S, float* __restrict__ nx) {
  __shared__ float sd[8192];
  __shared__ float rv[1024];
  __shared__ int   ri[1024];
  __shared__ float cc[3];
  __shared__ int   sfar;
  int b = blockIdx.x, t = threadIdx.x, T = blockDim.x;
  const float* xb = xyz + (size_t)b * N * 3;
  for (int j = t; j < N; j += T) sd[j] = 1e10f;
  if (t == 0) sfar = 0;
  __syncthreads();
  for (int s = 0; s < S; ++s) {
    int far = sfar;
    if (t < 3) cc[t] = xb[far * 3 + t];
    if (t == 0) {
      float* o = nx + ((size_t)b * S + s) * 3;
      o[0] = xb[far * 3 + 0]; o[1] = xb[far * 3 + 1]; o[2] = xb[far * 3 + 2];
    }
    __syncthreads();
    float bv = -1.0f; int bidx = 0x7fffffff;
    for (int j = t; j < N; j += T) {
      float dx = xb[j * 3 + 0] - cc[0];
      float dy = xb[j * 3 + 1] - cc[1];
      float dz = xb[j * 3 + 2] - cc[2];
      float d = dx * dx + dy * dy + dz * dz;
      float nd = fminf(sd[j], d);
      sd[j] = nd;
      if (nd > bv) { bv = nd; bidx = j; }
    }
    rv[t] = bv; ri[t] = bidx;
    __syncthreads();
    for (int off = T >> 1; off > 0; off >>= 1) {
      if (t < off) {
        float v2 = rv[t + off]; int i2 = ri[t + off];
        if (v2 > rv[t] || (v2 == rv[t] && i2 < ri[t])) { rv[t] = v2; ri[t] = i2; }
      }
      __syncthreads();
    }
    if (t == 0) sfar = ri[0];
    __syncthreads();
  }
}

// Ball query: first K indices (ascending) within radius, padded with first.
__global__ void k_ball(const float* __restrict__ xyz, int N,
                       const float* __restrict__ nx, int S,
                       float r2, int K, int* __restrict__ gidx) {
  extern __shared__ float sxyz[];
  int b = blockIdx.y;
  const float* xb = xyz + (size_t)b * N * 3;
  for (int i = threadIdx.x; i < N * 3; i += blockDim.x) sxyz[i] = xb[i];
  __syncthreads();
  int s = blockIdx.x * blockDim.x + threadIdx.x;
  if (s >= S) return;
  const float* c = nx + ((size_t)b * S + s) * 3;
  float cx = c[0], cy = c[1], cz = c[2];
  int* out = gidx + ((size_t)b * S + s) * K;
  int cnt = 0, first = 0;
  for (int j = 0; j < N && cnt < K; ++j) {
    float dx = sxyz[j * 3 + 0] - cx;
    float dy = sxyz[j * 3 + 1] - cy;
    float dz = sxyz[j * 3 + 2] - cz;
    float d = dx * dx + dy * dy + dz * dz;
    if (d <= r2) { if (cnt == 0) first = j; out[cnt++] = j; }
  }
  for (; cnt < K; ++cnt) out[cnt] = first;
}

// Build grouped features: [B][S*K][Cgs]; ch<[Cpv]: gathered feats,
// next 3: relative xyz, rest zero.
__global__ void k_group(const int* __restrict__ gidx, const h16* __restrict__ fprev,
                        int Cps, int Cpv, const float* __restrict__ xp,
                        const float* __restrict__ nx, h16* __restrict__ out,
                        int Cgs, int S, int K, int Np, int NB) {
  size_t total = (size_t)NB * S * K * Cgs;
  size_t stride = (size_t)gridDim.x * blockDim.x;
  for (size_t i = (size_t)blockIdx.x * blockDim.x + threadIdx.x; i < total; i += stride) {
    int c = (int)(i % Cgs); size_t r = i / Cgs;
    int k = (int)(r % K); size_t r2 = r / K;
    int s = (int)(r2 % S); int b = (int)(r2 / S);
    int idx = gidx[((size_t)b * S + s) * K + k];
    h16 v;
    if (c < Cpv) v = fprev[((size_t)b * Np + idx) * Cps + c];
    else if (c < Cpv + 3) {
      int d = c - Cpv;
      v = (h16)(xp[((size_t)b * Np + idx) * 3 + d] - nx[((size_t)b * S + s) * 3 + d]);
    } else v = (h16)0.0f;
    out[i] = v;
  }
}

// ---------------------------------------------------------------------------
// GEMM-support kernels
// ---------------------------------------------------------------------------
__global__ void k_pack_w(const float* __restrict__ w, h16* __restrict__ wh,
                         int M, int K, int Mp, int Kp) {
  int i = blockIdx.x * blockDim.x + threadIdx.x;
  if (i >= Mp * Kp) return;
  int c = i % Kp, r = i / Kp;
  wh[i] = (r < M && c < K) ? (h16)w[(size_t)r * K + c] : (h16)0.0f;
}

__global__ void k_prep_bn(const float* __restrict__ g, const float* __restrict__ bb,
                          const float* __restrict__ m, const float* __restrict__ v,
                          const float* __restrict__ cb,
                          float* __restrict__ sv, float* __restrict__ tv, int M, int Cp) {
  int c = blockIdx.x * blockDim.x + threadIdx.x;
  if (c >= Cp) return;
  if (c < M) {
    float s = g[c] * rsqrtf(v[c] + 1e-5f);
    sv[c] = s; tv[c] = (cb[c] - m[c]) * s + bb[c];
  } else { sv[c] = 0.0f; tv[c] = 0.0f; }
}

__global__ void k_prep_cond(const float* __restrict__ gw, const float* __restrict__ gb,
                            const float* __restrict__ bw, const float* __restrict__ cb,
                            const float* __restrict__ tin,
                            float* __restrict__ sv, float* __restrict__ tv,
                            int M, int Cp, int NB) {
  int i = blockIdx.x * blockDim.x + threadIdx.x;
  if (i >= NB * Cp) return;
  int c = i % Cp, b = i / Cp;
  if (c < M) {
    float tt = tin[b];
    float e0 = tt, e1 = sinf(tt), e2 = cosf(tt);
    float gz = gw[c * 3 + 0] * e0 + gw[c * 3 + 1] * e1 + gw[c * 3 + 2] * e2 + gb[c];
    float gate = 1.0f / (1.0f + expf(-gz));
    float bz = bw[c * 3 + 0] * e0 + bw[c * 3 + 1] * e1 + bw[c * 3 + 2] * e2;
    sv[i] = gate; tv[i] = cb[c] * gate + bz;
  } else { sv[i] = 0.0f; tv[i] = 0.0f; }
}

__global__ void k_prep_plain(const float* __restrict__ cb, float* __restrict__ sv,
                             float* __restrict__ tv, int M, int Cp) {
  int c = blockIdx.x * blockDim.x + threadIdx.x;
  if (c >= Cp) return;
  if (c < M) { sv[c] = 1.0f; tv[c] = cb[c]; }
  else       { sv[c] = 0.0f; tv[c] = 0.0f; }
}

// ---------------------------------------------------------------------------
// WMMA GEMM: Y[b][n][m] = relu?( (W x)[m][n]*s[m] + t[m] )
// W packed f16 [Mp][Kp]; X f16 [B][Nsp][Kp]; one wave = 16(M) x NSUB*16(N).
// NSUB is a compile-time constant so the hot loop is branch-free straight-line
// code. All fragments for a k-step are loaded before the WMMA group so the
// scheduler can clause the loads and overlap them with the XDL pipe.
// finalMode writes f32 to fOut[(b*Mvalid+m)*Nsp + n] for m<Mvalid.
// ---------------------------------------------------------------------------
template <int NSUB>
__global__ __launch_bounds__(256) void k_gemm(
    const h16* __restrict__ Wh, const h16* __restrict__ X, h16* __restrict__ Y,
    const float* __restrict__ svec, const float* __restrict__ tvec,
    int Mvalid, int Kp, int Nsp, int CoutStride, int mTiles,
    int nGroups, int nbase0,
    int stStride, int relu, int finalMode, float* __restrict__ fOut) {
  // wave id is wave-uniform by construction; force it into an SGPR so all
  // control flow below is scalar (s_cbranch, not exec masking).
  const int wlocal = __builtin_amdgcn_readfirstlane((int)(threadIdx.x >> 5));
  const int wid = (int)blockIdx.x * 8 + wlocal;
  if (wid >= mTiles * nGroups) return;            // scalar branch
  const int lane = threadIdx.x & 31;
  const int mt = wid % mTiles;
  const int ng = wid / mTiles;
  const int b  = blockIdx.y;

  const h16* Xb = X + (size_t)b * (size_t)Nsp * (size_t)Kp;
  const int mrow  = mt * 16 + (lane & 15);
  const int koff  = (lane >> 4) * 8;   // A-frag K offset is 2*koff (contiguous k = 16h + s)
  const int kboff = (lane >> 4) * 16;  // B-frag K offset
  const int coln  = lane & 15;
  const h16* wrow = Wh + (size_t)mrow * (size_t)Kp;
  const int nbase = nbase0 + ng * 64;

  v8f acc[NSUB];
  #pragma unroll
  for (int s = 0; s < NSUB; ++s)
    #pragma unroll
    for (int e = 0; e < 8; ++e) acc[s][e] = 0.0f;

  const h16* xr[NSUB];
  #pragma unroll
  for (int s = 0; s < NSUB; ++s)
    xr[s] = Xb + (size_t)(nbase + s * 16 + coln) * (size_t)Kp;

  for (int kt = 0; kt < Kp; kt += 32) {
    // ---- load phase: A-frag + all NSUB B-frags (clause-able, MLP) ----
    v8h alo = *(const v8h*)(wrow + kt + 2 * koff);
    v8h ahi = *(const v8h*)(wrow + kt + 2 * koff + 8);
    v8h blo[NSUB], bhi[NSUB];
    #pragma unroll
    for (int s = 0; s < NSUB; ++s) {
      __builtin_prefetch(xr[s] + kt + 64, 0, 1);   // global_prefetch_b8
      blo[s] = *(const v8h*)(xr[s] + kt + kboff);
      bhi[s] = *(const v8h*)(xr[s] + kt + kboff + 8);
    }
    // ---- compute phase: NSUB back-to-back WMMAs ----
    v16h a;
    #pragma unroll
    for (int e = 0; e < 8; ++e) { a[e] = alo[e]; a[8 + e] = ahi[e]; }
    #pragma unroll
    for (int s = 0; s < NSUB; ++s) {
      v16h bm;
      #pragma unroll
      for (int e = 0; e < 8; ++e) { bm[e] = blo[s][e]; bm[8 + e] = bhi[s][e]; }
      acc[s] = __builtin_amdgcn_wmma_f32_16x16x32_f16(false, a, false, bm,
                                                      (short)0, acc[s], false, false);
    }
  }

  const int m0 = mt * 16 + (lane >> 4) * 8;
  const float* sb = svec + (size_t)b * stStride;
  const float* tb = tvec + (size_t)b * stStride;
  v4f s0 = *(const v4f*)(sb + m0);
  v4f s1 = *(const v4f*)(sb + m0 + 4);
  v4f t0 = *(const v4f*)(tb + m0);
  v4f t1 = *(const v4f*)(tb + m0 + 4);
  #pragma unroll
  for (int s = 0; s < NSUB; ++s) {
    const int n = nbase + s * 16 + coln;
    float o[8];
    #pragma unroll
    for (int j = 0; j < 8; ++j) {
      float sc = (j < 4) ? s0[j] : s1[j - 4];
      float tc = (j < 4) ? t0[j] : t1[j - 4];
      float v = acc[s][j] * sc + tc;
      if (relu) v = fmaxf(v, 0.0f);
      if (m0 + j >= Mvalid) v = 0.0f;              // keep channel padding zero
      o[j] = v;
    }
    if (finalMode) {
      #pragma unroll
      for (int j = 0; j < 8; ++j) {
        int m = m0 + j;
        if (m < Mvalid) fOut[((size_t)b * Mvalid + m) * (size_t)Nsp + n] = o[j];
      }
    } else {
      v8h oh;
      #pragma unroll
      for (int j = 0; j < 8; ++j) oh[j] = (h16)o[j];
      *(v8h*)(Y + ((size_t)b * Nsp + n) * (size_t)CoutStride + m0) = oh;
    }
  }
}

// Max over the K group dim: out[b][s][coff+c] = max_k in[b][s*K+k][c]
__global__ void k_pool(const h16* __restrict__ in, h16* __restrict__ out,
                       int S, int K, int Cs, int Cb, int Co, int coff, int NB) {
  size_t total = (size_t)NB * S * Cb;
  size_t stride = (size_t)gridDim.x * blockDim.x;
  for (size_t i = (size_t)blockIdx.x * blockDim.x + threadIdx.x; i < total; i += stride) {
    int c = (int)(i % Cb);
    int s = (int)((i / Cb) % S);
    int b = (int)(i / ((size_t)Cb * S));
    float m = -3.4e38f;
    for (int k = 0; k < K; ++k) {
      float v = (float)in[(((size_t)b * S + s) * K + k) * Cs + c];
      m = fmaxf(m, v);
    }
    out[((size_t)b * S + s) * Co + coff + c] = (h16)m;
  }
}

// 3-NN + inverse-distance weights for feature propagation.
__global__ void k_fpknn(const float* __restrict__ x1, int N1,
                        const float* __restrict__ x2, int N2,
                        int* __restrict__ idx3, float* __restrict__ w3, int NB) {
  int i = blockIdx.x * blockDim.x + threadIdx.x;
  if (i >= NB * N1) return;
  int n = i % N1, b = i / N1;
  const float* p = x1 + ((size_t)b * N1 + n) * 3;
  const float* xb = x2 + (size_t)b * N2 * 3;
  float px = p[0], py = p[1], pz = p[2];
  float d0 = 1e30f, d1 = 1e30f, d2 = 1e30f;
  int i0 = 0, i1 = 0, i2 = 0;
  for (int j = 0; j < N2; ++j) {
    float dx = xb[j * 3 + 0] - px;
    float dy = xb[j * 3 + 1] - py;
    float dz = xb[j * 3 + 2] - pz;
    float d = dx * dx + dy * dy + dz * dz;
    if (d < d0)      { d2 = d1; i2 = i1; d1 = d0; i1 = i0; d0 = d; i0 = j; }
    else if (d < d1) { d2 = d1; i2 = i1; d1 = d; i1 = j; }
    else if (d < d2) { d2 = d; i2 = j; }
  }
  float w0 = 1.0f / (d0 + 1e-8f), w1 = 1.0f / (d1 + 1e-8f), w2 = 1.0f / (d2 + 1e-8f);
  float ws = w0 + w1 + w2;
  idx3[(size_t)i * 3 + 0] = i0; idx3[(size_t)i * 3 + 1] = i1; idx3[(size_t)i * 3 + 2] = i2;
  w3[(size_t)i * 3 + 0] = w0 / ws; w3[(size_t)i * 3 + 1] = w1 / ws; w3[(size_t)i * 3 + 2] = w2 / ws;
}

// Build FP input: [points1 | interp(points2)] padded.
__global__ void k_fpbuild(const h16* __restrict__ p1, int C1, int C1s,
                          const h16* __restrict__ p2, int C2, int C2s,
                          const int* __restrict__ idx3, const float* __restrict__ w3,
                          h16* __restrict__ out, int Cs, int N1, int N2, int NB) {
  size_t total = (size_t)NB * N1 * Cs;
  size_t stride = (size_t)gridDim.x * blockDim.x;
  for (size_t i = (size_t)blockIdx.x * blockDim.x + threadIdx.x; i < total; i += stride) {
    int c = (int)(i % Cs);
    int n = (int)((i / Cs) % N1);
    int b = (int)(i / ((size_t)Cs * N1));
    h16 v;
    if (c < C1) {
      v = p1[((size_t)b * N1 + n) * C1s + c];
    } else if (c < C1 + C2) {
      const int*   id = idx3 + ((size_t)b * N1 + n) * 3;
      const float* w  = w3   + ((size_t)b * N1 + n) * 3;
      int cc = c - C1;
      float acc = 0.0f;
      for (int j = 0; j < 3; ++j)
        acc += w[j] * (float)p2[((size_t)b * N2 + id[j]) * C2s + cc];
      v = (h16)acc;
    } else v = (h16)0.0f;
    out[i] = v;
  }
}

// ---------------------------------------------------------------------------
// Host driver
// ---------------------------------------------------------------------------
static inline int align32h(int x) { return (x + 31) & ~31; }

extern "C" void kernel_launch(void* const* d_in, const int* in_sizes, int n_in,
                              void* d_out, int out_size, void* d_ws, size_t ws_size,
                              hipStream_t stream) {
  (void)in_sizes; (void)n_in; (void)out_size; (void)ws_size;
  const int Bn = 2;
  static const int LVL_N[5]  = {8192, 1024, 256, 64, 16};
  static const int FEAT_CS[5]= {64, 96, 256, 512, 1024};
  static const int LVL_CV[5] = {48, 96, 256, 512, 1024};
  static const float SA_R[4][2] = {{0.05f,0.1f},{0.1f,0.2f},{0.2f,0.4f},{0.4f,0.8f}};
  static const int   SA_K[4][2] = {{16,32},{16,32},{16,32},{16,32}};
  static const int SA_MLP[4][2][3] = {
      {{16,16,32},{32,32,64}}, {{64,64,128},{64,96,128}},
      {{128,196,256},{128,196,256}}, {{256,256,512},{256,384,512}}};
  static const int COND_D[7] = {96, 256, 512, 1024, 256, 256, 128};

  // Param leaf indexing (flattened pytree, sorted dict keys):
  //   d_in[0]=query_pts [1]=mid_pts(unused) [2]=t
  //   cond[i]: base 3+5i : bias_w, gate_b, gate_w, layer.b, layer.w
  //   fp layer j: base 38+6j : bn.b, bn.g, bn.m, bn.v, conv.b, conv.w
  //   head: 92 c1.b, 93 c1.w, 94 c2.b, 95 c2.w
  //   sa layer j=(st*6+br*3+l): base 96+6j : bn.b, bn.g, bn.m, bn.v, conv.b, conv.w
  auto P = [&](int i) { return (const float*)d_in[i]; };
  const float* tptr = (const float*)d_in[2];

  // --- workspace bump allocator ---
  char* wsb = (char*)d_ws;
  size_t wo = 0;
  auto bump = [&](size_t bytes) -> void* {
    size_t a = (wo + 255) & ~(size_t)255;
    wo = a + bytes;
    return (void*)(wsb + a);
  };

  float* xyzL[5];
  for (int l = 0; l < 5; ++l) xyzL[l] = (float*)bump((size_t)Bn * LVL_N[l] * 3 * 4);
  h16* featL[5];
  for (int l = 0; l < 5; ++l) featL[l] = (h16*)bump((size_t)Bn * LVL_N[l] * FEAT_CS[l] * 2);
  h16* fpo[3];
  fpo[0] = (h16*)bump((size_t)Bn * 64 * 256 * 2);
  fpo[1] = (h16*)bump((size_t)Bn * 256 * 256 * 2);
  fpo[2] = (h16*)bump((size_t)Bn * 1024 * 128 * 2);
  const size_t BUFB = 9u * 1024u * 1024u;
  h16* bufA = (h16*)bump(BUFB);
  h16* bufB = (h16*)bump(BUFB);
  int* gidx = (int*)bump((size_t)Bn * 1024 * 32 * 4);
  int* idx3 = (int*)bump((size_t)Bn * 8192 * 3 * 4);
  float* w3 = (float*)bump((size_t)Bn * 8192 * 3 * 4);

  auto gsBlocks = [](size_t total) {
    long b = (long)((total + 255) / 256);
    return (int)(b < 16384 ? b : 16384);
  };

  auto packW = [&](const float* w, int M, int K, int Kp) -> h16* {
    int Mp = align32h(M);
    h16* wh = (h16*)bump((size_t)Mp * Kp * 2);
    int tot = Mp * Kp;
    k_pack_w<<<(tot + 255) / 256, 256, 0, stream>>>(w, wh, M, K, Mp, Kp);
    return wh;
  };

  // Split Nsp into full 64-column groups (branch-free NSUB=4 kernel) plus at
  // most one 16/32/48-column tail launch.
  auto gemmL = [&](const h16* Wh, int Kp, const h16* X, int Nsp, h16* Y,
                   int CoutStride, int mTiles, int Mvalid,
                   const float* sv, const float* tv, int stStride,
                   int relu, int fin, float* fOut) {
    int nFull = Nsp / 64;
    int tail  = (Nsp % 64) / 16;
    if (nFull > 0) {
      int waves = mTiles * nFull;
      int blocks = (waves + 7) / 8;
      k_gemm<4><<<dim3(blocks, Bn), 256, 0, stream>>>(Wh, X, Y, sv, tv, Mvalid, Kp, Nsp,
                                                      CoutStride, mTiles, nFull, 0,
                                                      stStride, relu, fin, fOut);
    }
    if (tail > 0) {
      int blocks = (mTiles + 7) / 8;
      int nb0 = nFull * 64;
      switch (tail) {
        case 1: k_gemm<1><<<dim3(blocks, Bn), 256, 0, stream>>>(Wh, X, Y, sv, tv, Mvalid, Kp, Nsp,
                    CoutStride, mTiles, 1, nb0, stStride, relu, fin, fOut); break;
        case 2: k_gemm<2><<<dim3(blocks, Bn), 256, 0, stream>>>(Wh, X, Y, sv, tv, Mvalid, Kp, Nsp,
                    CoutStride, mTiles, 1, nb0, stStride, relu, fin, fOut); break;
        default: k_gemm<3><<<dim3(blocks, Bn), 256, 0, stream>>>(Wh, X, Y, sv, tv, Mvalid, Kp, Nsp,
                    CoutStride, mTiles, 1, nb0, stStride, relu, fin, fOut); break;
      }
    }
  };

  // conv + BN + ReLU layer (leafBase -> bn.b,bn.g,bn.m,bn.v,conv.b,conv.w)
  auto bnGemm = [&](int leafBase, int M, int Cv, int CinP,
                    const h16* in, int Nsp, h16* out) -> int {
    int Mp = align32h(M);
    h16* Wh = packW(P(leafBase + 5), M, Cv, CinP);
    float* sv = (float*)bump((size_t)Mp * 4);
    float* tv = (float*)bump((size_t)Mp * 4);
    k_prep_bn<<<(Mp + 255) / 256, 256, 0, stream>>>(P(leafBase + 1), P(leafBase + 0),
                                                    P(leafBase + 2), P(leafBase + 3),
                                                    P(leafBase + 4), sv, tv, M, Mp);
    gemmL(Wh, CinP, in, Nsp, out, Mp, Mp / 16, M, sv, tv, 0, 1, 0, nullptr);
    return Mp;
  };

  // time-conditioned conv: y = (Wx+b)*gate + bias
  auto condGemm = [&](int ci, const h16* X, int Nsp, h16* Y) {
    int dim = COND_D[ci];
    int cb = 3 + 5 * ci;
    h16* Wh = packW(P(cb + 4), dim, dim, dim);
    float* sv = (float*)bump((size_t)Bn * dim * 4);
    float* tv = (float*)bump((size_t)Bn * dim * 4);
    int tot = Bn * dim;
    k_prep_cond<<<(tot + 255) / 256, 256, 0, stream>>>(P(cb + 2), P(cb + 1), P(cb + 0),
                                                       P(cb + 3), tptr, sv, tv, dim, dim, Bn);
    gemmL(Wh, dim, X, Nsp, Y, dim, dim / 16, dim, sv, tv, dim, 0, 0, nullptr);
  };

  // ---- stage 0 inputs ----
  k_xyz_t<<<dim3((LVL_N[0] + 255) / 256, Bn), 256, 0, stream>>>((const float*)d_in[0], xyzL[0], LVL_N[0]);
  k_knn<<<dim3(LVL_N[0] / 256, Bn), 256, LVL_N[0] * 12, stream>>>(xyzL[0], featL[0], LVL_N[0], FEAT_CS[0]);

  // ---- set abstraction stages ----
  for (int st = 0; st < 4; ++st) {
    int Nprev = LVL_N[st], S = LVL_N[st + 1];
    int Cprev = LVL_CV[st];
    k_fps<<<Bn, 1024, 0, stream>>>(xyzL[st], Nprev, S, xyzL[st + 1]);
    int coff = 0;
    for (int br = 0; br < 2; ++br) {
      float r = SA_R[st][br];
      int K = SA_K[st][br];
      k_ball<<<dim3((S + 255) / 256, Bn), 256, Nprev * 12, stream>>>(
          xyzL[st], Nprev, xyzL[st + 1], S, r * r, K, gidx);
      int Cg = Cprev + 3, Cgp = align32h(Cg);
      size_t gtot = (size_t)Bn * S * K * Cgp;
      k_group<<<gsBlocks(gtot), 256, 0, stream>>>(gidx, featL[st], FEAT_CS[st], Cprev,
                                                  xyzL[st], xyzL[st + 1], bufA, Cgp, S, K, Nprev, Bn);
      h16* in = bufA; h16* out = bufB;
      int Cv = Cg, CinP = Cgp;
      for (int l = 0; l < 3; ++l) {
        int M = SA_MLP[st][br][l];
        int base = 96 + 6 * (st * 6 + br * 3 + l);
        int Mp = bnGemm(base, M, Cv, CinP, in, S * K, out);
        h16* t2 = in; in = out; out = t2;
        Cv = M; CinP = Mp;
      }
      size_t ptot = (size_t)Bn * S * Cv;
      k_pool<<<gsBlocks(ptot), 256, 0, stream>>>(in, featL[st + 1], S, K, CinP, Cv,
                                                 FEAT_CS[st + 1], coff, Bn);
      coff += Cv;
    }
    condGemm(st, featL[st + 1], S, bufA);
    hipMemcpyAsync(featL[st + 1], bufA, (size_t)Bn * S * FEAT_CS[st + 1] * 2,
                   hipMemcpyDeviceToDevice, stream);
  }

  // ---- feature propagation stages ----
  const int fpUp[4] = {3, 2, 1, 0};
  const int fpDn[4] = {4, 3, 2, 1};
  const int fpC1[4] = {512, 256, 96, 0};
  const h16* fpP1[4] = {featL[3], featL[2], featL[1], nullptr};
  const int fpC2[4] = {1024, 256, 256, 128};
  const h16* fpP2[4] = {featL[4], fpo[0], fpo[1], fpo[2]};
  const int fpLs[4] = {0, 2, 4, 6};
  const int fpNL[4] = {2, 2, 2, 3};
  const int fpCh[4][4] = {{1536,256,256,0},{512,256,256,0},{352,256,128,0},{128,128,128,128}};
  h16* fpLast = nullptr;

  for (int f = 0; f < 4; ++f) {
    int u = fpUp[f], d = fpDn[f];
    int N1 = LVL_N[u], N2 = LVL_N[d];
    int tot1 = Bn * N1;
    k_fpknn<<<(tot1 + 255) / 256, 256, 0, stream>>>(xyzL[u], N1, xyzL[d], N2, idx3, w3, Bn);
    int C1 = fpC1[f], C2 = fpC2[f], Cin = C1 + C2;  // all multiples of 32
    size_t btot = (size_t)Bn * N1 * Cin;
    k_fpbuild<<<gsBlocks(btot), 256, 0, stream>>>(fpP1[f], C1, (C1 > 0 ? C1 : 1),
                                                  fpP2[f], C2, fpC2[f],
                                                  idx3, w3, bufA, Cin, N1, N2, Bn);
    h16* in = bufA; h16* out = bufB;
    int Cv = Cin, CinP = Cin;
    for (int l = 0; l < fpNL[f]; ++l) {
      int base = 38 + 6 * (fpLs[f] + l);
      int M = fpCh[f][l + 1];
      int Mp = bnGemm(base, M, Cv, CinP, in, N1, out);
      h16* t2 = in; in = out; out = t2;
      Cv = M; CinP = Mp;
    }
    if (f < 3) condGemm(4 + f, in, N1, fpo[f]);
    else       fpLast = in;   // l0 features, 128ch @ 8192 pts
  }

  // ---- head ----
  {
    h16* Wh1 = packW(P(93), 128, 128, 128);
    float* sv1 = (float*)bump(128 * 4);
    float* tv1 = (float*)bump(128 * 4);
    k_prep_plain<<<1, 256, 0, stream>>>(P(92), sv1, tv1, 128, 128);
    h16* hout = (fpLast == bufA) ? bufB : bufA;
    gemmL(Wh1, 128, fpLast, 8192, hout, 128, 8, 128, sv1, tv1, 0, 1, 0, nullptr);

    h16* Wh2 = packW(P(95), 3, 128, 128);
    float* sv2 = (float*)bump(16 * 4);
    float* tv2 = (float*)bump(16 * 4);
    k_prep_plain<<<1, 256, 0, stream>>>(P(94), sv2, tv2, 3, 16);
    gemmL(Wh2, 128, hout, 8192, nullptr, 16, 1, 3, sv2, tv2, 0, 0, 1, (float*)d_out);
  }
}
